// NDPNoisePredictor_24842090840465
// MI455X (gfx1250) — matrix operand order repaired and checked by reading, not compiled, not run on hardware
//
#include <hip/hip_runtime.h>
#include <hip/hip_bf16.h>

// ---------------------------------------------------------------------------
// Problem constants (from reference): B=8192, IN=64, H=512, L=6, NH=8
// Seq dims are 1 => attention collapses to V-proj + O-proj GEMMs.
// ---------------------------------------------------------------------------
#define BB   8192
#define INF_ 64
#define HD   512
#define LL   6
#define H4   2048

typedef __bf16 bf16;
typedef __attribute__((ext_vector_type(16))) __bf16 v16bf;
typedef __attribute__((ext_vector_type(8)))  float  v8f;

union Frag {
    v16bf    v;
    unsigned u[8];
};

__device__ __forceinline__ float gelu_f(float x) {
    // exact GELU: 0.5*x*(1+erf(x/sqrt(2)))
    return 0.5f * x * (1.0f + erff(x * 0.70710678118654752440f));
}

// gfx1250 async copy: global -> LDS, 16B per lane, tracked by ASYNCcnt.
// vdst = VGPR holding LDS byte address, vaddr = 64-bit global address.
__device__ __forceinline__ void async_ld_b128(unsigned lds_byte_addr, const bf16* gptr) {
    asm volatile("global_load_async_to_lds_b128 %0, %1, off"
                 :: "v"(lds_byte_addr), "v"(gptr)
                 : "memory");
}

// ---------------------------------------------------------------------------
// fp32 -> bf16 convert
// ---------------------------------------------------------------------------
__global__ void f2b_kernel(const float* __restrict__ src, bf16* __restrict__ dst, int n) {
    int i = blockIdx.x * blockDim.x + threadIdx.x;
    if (i < n) dst[i] = (bf16)src[i];
}

// ---------------------------------------------------------------------------
// y path: out[b,n] = gelu(y[b]*w1[n] + b1[n])  (K=1 linear, fused gelu, bf16 out)
// ---------------------------------------------------------------------------
__global__ void ypath_kernel(const float* __restrict__ y, const float* __restrict__ w1,
                             const float* __restrict__ b1, bf16* __restrict__ out, int total) {
    int i = blockIdx.x * blockDim.x + threadIdx.x;
    if (i >= total) return;
    int b = i >> 9;          // /512
    int n = i & 511;
    out[i] = (bf16)gelu_f(y[b] * w1[n] + b1[n]);
}

// ---------------------------------------------------------------------------
// sinusoidal timestep embedding, half=256: te[b, i<256]=sin(t*f_i), else cos
// ---------------------------------------------------------------------------
__global__ void time_embed_kernel(const int* __restrict__ t, bf16* __restrict__ te, int total) {
    int i = blockIdx.x * blockDim.x + threadIdx.x;
    if (i >= total) return;
    int b = i >> 9;
    int n = i & 511;
    int k = (n < 256) ? n : (n - 256);
    float f = expf((float)k * (-9.210340371976184f / 256.0f)); // -ln(10000)/half
    float e = (float)t[b] * f;
    te[i] = (bf16)((n < 256) ? sinf(e) : cosf(e));
}

// ---------------------------------------------------------------------------
// LayerNorm over H=512 per row, one wave32 per row, bf16 output, optional gelu
// ---------------------------------------------------------------------------
__global__ __launch_bounds__(256)
void ln_bf16_kernel(const float* __restrict__ x, const float* __restrict__ w,
                    const float* __restrict__ b, bf16* __restrict__ out, int post_gelu) {
    int row  = blockIdx.x * 8 + (threadIdx.x >> 5);
    int lane = threadIdx.x & 31;
    const float* xr = x + (size_t)row * HD;
    float v[16];
    float s = 0.f, ss = 0.f;
#pragma unroll
    for (int j = 0; j < 16; ++j) {
        float t = xr[lane * 16 + j];
        v[j] = t; s += t; ss += t * t;
    }
#pragma unroll
    for (int m = 16; m >= 1; m >>= 1) {   // wave32 reduction
        s  += __shfl_xor(s,  m, 32);
        ss += __shfl_xor(ss, m, 32);
    }
    float mean = s * (1.0f / HD);
    float var  = ss * (1.0f / HD) - mean * mean;
    float r    = rsqrtf(var + 1e-5f);
    bf16* outr = out + (size_t)row * HD;
#pragma unroll
    for (int j = 0; j < 16; ++j) {
        int c = lane * 16 + j;
        float t = (v[j] - mean) * r * w[c] + b[c];
        if (post_gelu) t = gelu_f(t);
        outr[c] = (bf16)t;
    }
}

// ---------------------------------------------------------------------------
// Final N=1 projection: out[b] = dot(g[b,:], m2w) + m2b  (wave per row)
// ---------------------------------------------------------------------------
__global__ __launch_bounds__(256)
void final_dot_kernel(const bf16* __restrict__ g, const float* __restrict__ w,
                      const float* __restrict__ bias, float* __restrict__ out) {
    int row  = blockIdx.x * 8 + (threadIdx.x >> 5);
    int lane = threadIdx.x & 31;
    const bf16* gr = g + (size_t)row * HD;
    float s = 0.f;
#pragma unroll
    for (int j = 0; j < 16; ++j) {
        int c = lane * 16 + j;
        s += (float)gr[c] * w[c];
    }
#pragma unroll
    for (int m = 16; m >= 1; m >>= 1) s += __shfl_xor(s, m, 32);
    if (lane == 0) out[row] = s + bias[0];
}

// ---------------------------------------------------------------------------
// WMMA bf16 GEMM:  out[M,N] = X[M,K] @ W[N,K]^T (+bias +res1 +res2, opt gelu)
// Block: 256 threads (8 waves). Tile: BM=128 x BN=64, BK=32.
// Double-buffered LDS, staged with global_load_async_to_lds_b128 (ASYNCcnt),
// issued two K-tiles ahead so HBM latency overlaps the v_wmma stream.
// Wave w owns rows [16w,16w+16) of the tile x all 64 cols -> 4 v_wmma per K-step.
// ---------------------------------------------------------------------------
#define BM 128
#define BN 64
#define BK 32
#define LDST (BK + 8)   // LDS stride in bf16 elements (padded vs 64 banks)

__global__ __launch_bounds__(256)
void gemm_bf16_kernel(const bf16* __restrict__ X, const bf16* __restrict__ W,
                      const float* __restrict__ bias,
                      const float* __restrict__ res1, const float* __restrict__ res2,
                      float* __restrict__ outF, bf16* __restrict__ outB,
                      int M, int N, int K, int gelu_flag) {
    __shared__ bf16 As[2][BM * LDST];
    __shared__ bf16 Bs[2][BN * LDST];

    const int tid   = threadIdx.x;
    const int wave  = tid >> 5;
    const int lane  = tid & 31;
    const int mBase = blockIdx.y * BM;
    const int nBase = blockIdx.x * BN;

    // per-thread staging assignment (3 x b128 async copies per K-tile)
    const int ar  = tid >> 1;            // A tile row 0..127
    const int acb = (tid & 1) * 16;      // A tile col 0 / 16
    const int br  = tid >> 2;            // B tile row 0..63
    const int bcb = (tid & 3) * 8;       // B tile col 0/8/16/24
    const bf16* aSrc = X + (size_t)(mBase + ar) * K + acb;
    const bf16* bSrc = W + (size_t)(nBase + br) * K + bcb;

    auto issue = [&](int k0, int p) {
        async_ld_b128((unsigned)(size_t)&As[p][ar * LDST + acb],     aSrc + k0);
        async_ld_b128((unsigned)(size_t)&As[p][ar * LDST + acb + 8], aSrc + k0 + 8);
        async_ld_b128((unsigned)(size_t)&Bs[p][br * LDST + bcb],     bSrc + k0);
    };

    v8f acc[4];
#pragma unroll
    for (int c = 0; c < 4; ++c)
#pragma unroll
        for (int j = 0; j < 8; ++j) acc[c][j] = 0.0f;

    const int nt = K / BK;
    issue(0, 0);
    if (nt > 1) issue(BK, 1);

    for (int t = 0; t < nt; ++t) {
        // wait for tile t's 3 copies; leave tile t+1's 3 in flight
        if (t + 1 < nt) asm volatile("s_wait_asynccnt 0x3" ::: "memory");
        else            asm volatile("s_wait_asynccnt 0x0" ::: "memory");
        __syncthreads();                     // tile t visible block-wide

        const int p = t & 1;
        // ---- A fragment (contiguous k order, same as B):
        // lanes 0-15: M=lane,   K 0..15
        // lanes16-31: M=lane-16,K 16..31
        Frag a;
        {
            int r     = (lane & 15) + wave * 16;
            int khalf = (lane >> 4) * 16;
            const unsigned* pp = (const unsigned*)&As[p][r * LDST];
#pragma unroll
            for (int j = 0; j < 4; ++j) a.u[j]     = pp[(khalf >> 1) + j];
#pragma unroll
            for (int j = 0; j < 4; ++j) a.u[4 + j] = pp[((khalf + 8) >> 1) + j];
        }
        // ---- 4 column sub-tiles: B fragment (32x16) + WMMA ----
#pragma unroll
        for (int c = 0; c < 4; ++c) {
            Frag b;
            int n  = (lane & 15) + c * 16;
            int kh = (lane >> 4) * 16;   // lanes 0-15: K 0..15 ; lanes 16-31: K 16..31
            const unsigned* pp = (const unsigned*)&Bs[p][n * LDST + kh];
#pragma unroll
            for (int j = 0; j < 8; ++j) b.u[j] = pp[j];
            acc[c] = __builtin_amdgcn_wmma_f32_16x16x32_bf16(
                false, a.v, false, b.v, (short)0, acc[c], false, false);
        }
        __syncthreads();                     // all waves done with buffer p
        if (t + 2 < nt) issue((t + 2) * BK, p);   // refill freed buffer
    }

    // ---- epilogue: C/D layout lanes 0-15 N=lane M=j; lanes 16-31 N=lane-16 M=8+j
    const int mw = mBase + wave * 16 + (lane >> 4) * 8;
#pragma unroll
    for (int c = 0; c < 4; ++c) {
        int n = nBase + c * 16 + (lane & 15);
        float bv = bias ? bias[n] : 0.0f;
#pragma unroll
        for (int j = 0; j < 8; ++j) {
            size_t idx = (size_t)(mw + j) * N + n;
            float v = acc[c][j] + bv;
            if (res1) v += res1[idx];
            if (res2) v += res2[idx];
            if (gelu_flag) v = gelu_f(v);
            if (outF) outF[idx] = v;
            if (outB) outB[idx] = (bf16)v;
        }
    }
}

// ---------------------------------------------------------------------------
// Host-side orchestration
// ---------------------------------------------------------------------------
extern "C" void kernel_launch(void* const* d_in, const int* in_sizes, int n_in,
                              void* d_out, int out_size, void* d_ws, size_t ws_size,
                              hipStream_t stream) {
    (void)in_sizes; (void)n_in; (void)out_size; (void)ws_size;

    const float* x       = (const float*)d_in[0];
    const float* y_noisy = (const float*)d_in[1];
    const float* xp_w1   = (const float*)d_in[2];
    const float* xp_b1   = (const float*)d_in[3];
    const float* xp_w2   = (const float*)d_in[4];
    const float* xp_b2   = (const float*)d_in[5];
    const float* yp_w1   = (const float*)d_in[6];
    const float* yp_b1   = (const float*)d_in[7];
    const float* yp_w2   = (const float*)d_in[8];
    const float* yp_b2   = (const float*)d_in[9];
    const float* td_w1   = (const float*)d_in[10];
    const float* td_b1   = (const float*)d_in[11];
    const float* td_w2   = (const float*)d_in[12];
    const float* td_b2   = (const float*)d_in[13];
    const float* n1_w    = (const float*)d_in[14];
    const float* n1_b    = (const float*)d_in[15];
    const float* n2_w    = (const float*)d_in[16];
    const float* n2_b    = (const float*)d_in[17];
    const float* ad_wqkv = (const float*)d_in[18];
    const float* ad_bqkv = (const float*)d_in[19];
    const float* ad_wo   = (const float*)d_in[20];
    const float* ad_bo   = (const float*)d_in[21];
    const float* as_wqkv = (const float*)d_in[22];
    const float* as_bqkv = (const float*)d_in[23];
    const float* as_wo   = (const float*)d_in[24];
    const float* as_bo   = (const float*)d_in[25];
    const float* fn_w    = (const float*)d_in[26];
    const float* fn_b    = (const float*)d_in[27];
    const float* f1_w    = (const float*)d_in[28];
    const float* f1_b    = (const float*)d_in[29];
    const float* f2_w    = (const float*)d_in[30];
    const float* f2_b    = (const float*)d_in[31];
    const float* pn_w    = (const float*)d_in[32];
    const float* pn_b    = (const float*)d_in[33];
    const float* m1_w    = (const float*)d_in[34];
    const float* m1_b    = (const float*)d_in[35];
    const float* m2_w    = (const float*)d_in[36];
    const float* m2_b    = (const float*)d_in[37];
    const int*   tsteps  = (const int*)d_in[38];

    const int HH  = HD * HD;        // 262144
    const int HH4 = H4 * HD;        // 1048576

    // bump allocator over workspace
    char*  base = (char*)d_ws;
    size_t off  = 0;
    auto alloc = [&](size_t bytes) -> void* {
        void* r = base + off;
        off = (off + bytes + 255) & ~(size_t)255;
        return r;
    };

    // bf16 weight copies
    bf16* w_xp1 = (bf16*)alloc((size_t)HD * INF_ * 2);
    bf16* w_xp2 = (bf16*)alloc((size_t)HH * 2);
    bf16* w_yp2 = (bf16*)alloc((size_t)HH * 2);
    bf16* w_td1 = (bf16*)alloc((size_t)HH * 2);
    bf16* w_td2 = (bf16*)alloc((size_t)HH * 2);
    bf16* w_adv = (bf16*)alloc((size_t)LL * HH * 2);
    bf16* w_ado = (bf16*)alloc((size_t)LL * HH * 2);
    bf16* w_asv = (bf16*)alloc((size_t)LL * HH * 2);
    bf16* w_aso = (bf16*)alloc((size_t)LL * HH * 2);
    bf16* w_f1  = (bf16*)alloc((size_t)LL * HH4 * 2);
    bf16* w_f2  = (bf16*)alloc((size_t)LL * HH4 * 2);
    bf16* w_m1  = (bf16*)alloc((size_t)HH * 2);

    // activations
    bf16*  x_bf = (bf16*)alloc((size_t)BB * INF_ * 2);
    bf16*  abuf = (bf16*)alloc((size_t)BB * HD * 2);
    bf16*  vbuf = (bf16*)alloc((size_t)BB * HD * 2);
    bf16*  ubuf = (bf16*)alloc((size_t)BB * H4 * 2);
    float* hF   = (float*)alloc((size_t)BB * HD * 4);
    float* aF   = (float*)alloc((size_t)BB * HD * 4);
    float* bF   = (float*)alloc((size_t)BB * HD * 4);

    auto conv = [&](const float* s, bf16* d, int n) {
        f2b_kernel<<<(n + 255) / 256, 256, 0, stream>>>(s, d, n);
    };
    auto gemm = [&](const bf16* Xp, const bf16* Wp, const float* bias,
                    const float* r1, const float* r2, float* oF, bf16* oB,
                    int M, int N, int K, int gl) {
        dim3 grid(N / BN, M / BM);
        gemm_bf16_kernel<<<grid, 256, 0, stream>>>(Xp, Wp, bias, r1, r2, oF, oB, M, N, K, gl);
    };
    auto ln = [&](const float* src, const float* w, const float* b, bf16* dst, int pg) {
        ln_bf16_kernel<<<BB / 8, 256, 0, stream>>>(src, w, b, dst, pg);
    };

    // --- weight conversion (V-slice of wqkv is all attention needs: seq len 1) ---
    conv(xp_w1, w_xp1, HD * INF_);
    conv(xp_w2, w_xp2, HH);
    conv(yp_w2, w_yp2, HH);
    conv(td_w1, w_td1, HH);
    conv(td_w2, w_td2, HH);
    conv(m1_w,  w_m1,  HH);
    for (int l = 0; l < LL; ++l) {
        conv(ad_wqkv + (size_t)l * 3 * HH + 2 * HH, w_adv + (size_t)l * HH, HH);
        conv(as_wqkv + (size_t)l * 3 * HH + 2 * HH, w_asv + (size_t)l * HH, HH);
        conv(ad_wo + (size_t)l * HH, w_ado + (size_t)l * HH, HH);
        conv(as_wo + (size_t)l * HH, w_aso + (size_t)l * HH, HH);
        conv(f1_w + (size_t)l * HH4, w_f1 + (size_t)l * HH4, HH4);
        conv(f2_w + (size_t)l * HH4, w_f2 + (size_t)l * HH4, HH4);
    }

    // --- head: h = h_x + h_y + time ---
    conv(x, x_bf, BB * INF_);
    gemm(x_bf, w_xp1, xp_b1, nullptr, nullptr, nullptr, abuf, BB, HD, INF_, 1); // gelu(x@w1)
    gemm(abuf, w_xp2, xp_b2, nullptr, nullptr, aF, nullptr, BB, HD, HD, 0);     // h_x -> aF

    ypath_kernel<<<(BB * HD + 255) / 256, 256, 0, stream>>>(y_noisy, yp_w1, yp_b1, vbuf, BB * HD);
    gemm(vbuf, w_yp2, yp_b2, aF, nullptr, bF, nullptr, BB, HD, HD, 0);          // h_x+h_y -> bF

    time_embed_kernel<<<(BB * HD + 255) / 256, 256, 0, stream>>>(tsteps, abuf, BB * HD);
    gemm(abuf, w_td1, td_b1, nullptr, nullptr, nullptr, vbuf, BB, HD, HD, 1);   // gelu(te@td1)
    gemm(vbuf, w_td2, td_b2, bF, nullptr, hF, nullptr, BB, HD, HD, 0);          // h -> hF

    // --- transformer layers (seq=1 => attention = V-proj then O-proj) ---
    for (int l = 0; l < LL; ++l) {
        const float* bvd = ad_bqkv + (size_t)l * 3 * HD + 2 * HD;
        const float* bvs = as_bqkv + (size_t)l * 3 * HD + 2 * HD;

        ln(hF, n1_w + (size_t)l * HD, n1_b + (size_t)l * HD, abuf, 0);
        gemm(abuf, w_adv + (size_t)l * HH, bvd, nullptr, nullptr, nullptr, vbuf, BB, HD, HD, 0);
        gemm(vbuf, w_ado + (size_t)l * HH, ad_bo + (size_t)l * HD,
             nullptr, nullptr, aF, nullptr, BB, HD, HD, 0);                      // d_out -> aF

        ln(hF, n2_w + (size_t)l * HD, n2_b + (size_t)l * HD, abuf, 0);
        gemm(abuf, w_asv + (size_t)l * HH, bvs, nullptr, nullptr, nullptr, vbuf, BB, HD, HD, 0);
        gemm(vbuf, w_aso + (size_t)l * HH, as_bo + (size_t)l * HD,
             hF, aF, bF, nullptr, BB, HD, HD, 0);                                // out = h+d+n -> bF

        ln(bF, fn_w + (size_t)l * HD, fn_b + (size_t)l * HD, abuf, 0);
        gemm(abuf, w_f1 + (size_t)l * HH4, f1_b + (size_t)l * H4,
             nullptr, nullptr, nullptr, ubuf, BB, H4, HD, 1);                    // gelu(f1) -> ubuf
        gemm(ubuf, w_f2 + (size_t)l * HH4, f2_b + (size_t)l * HD,
             bF, nullptr, hF, nullptr, BB, HD, H4, 0);                           // h = out + f2(..)
    }

    // --- tail: y = m2 @ gelu(m1 @ gelu(ln(h))) ---
    ln(hF, pn_w, pn_b, abuf, 1);                                                 // gelu(ln(h))
    gemm(abuf, w_m1, m1_b, nullptr, nullptr, nullptr, vbuf, BB, HD, HD, 1);      // gelu(m1..)
    final_dot_kernel<<<BB / 8, 256, 0, stream>>>(vbuf, m2_w, m2_b, (float*)d_out);
}
